// SpatialFeatureGenerator_38774964749009
// MI455X (gfx1250) — compile-verified, matrix operand order checked by reading, not run on hardware
//
#include <hip/hip_runtime.h>

// ---------------------------------------------------------------------------
// SpatialFeatureGenerator for MI455X (gfx1250, wave32, WMMA).
//   K1 raster+bilinear-resize -> maps f32 [256][2][100][100]
//   K2 conv1 (2->16, VALU)    -> act1 bf16 [256][10000][16]   (pixel-major)
//   K3 pack conv2 weights into WMMA B-fragment order (bf16)
//   K4 conv2 as implicit-GEMM WMMA bf16 -> act2 bf16 [256][32*10000] (CHW flat)
//   K5 proj_w f32 -> bf16
//   K6 projection GEMM (256x256x320000) WMMA bf16, K split 16, B panel staged
//      in LDS via global_load_async_to_lds_b128 (double buffered, ASYNCcnt)
//   K7 reduce partials + bias -> d_out f32 (32,8,256)
// ---------------------------------------------------------------------------

#define NSAMP   256
#define GRID_W  100
#define NPIX    (GRID_W * GRID_W)      // 10000
#define C1CH    16
#define C2CH    32
#define FDIM    256
#define KPROJ   (C2CH * NPIX)          // 320000
#define HUMANSZ 100
#define KSPLIT  16
#define KCHUNK  (KPROJ / KSPLIT)       // 20000
#define KSTEPS  (KCHUNK / 32)          // 625
#define PIXTILES (NPIX / 16)           // 625 pixel tiles per sample

typedef __bf16 bf16;
typedef __attribute__((ext_vector_type(16))) __bf16 v16bf;
typedef __attribute__((ext_vector_type(8)))  __bf16 bf8;
typedef __attribute__((ext_vector_type(4)))  __bf16 bf4;
typedef __attribute__((ext_vector_type(4)))  float  f4;
typedef __attribute__((ext_vector_type(8)))  float  v8f;

// gcc-style vector: exact pointee type of the async-to-LDS builtin's pointer
// params (clang printed 'vector_size(4*sizeof(int)) int __device__ *').
typedef int async_v4i __attribute__((vector_size(16)));

#ifndef __has_builtin
#define __has_builtin(x) 0
#endif
#if __has_builtin(__builtin_amdgcn_global_load_async_to_lds_b128) && \
    __has_builtin(__builtin_amdgcn_s_wait_asynccnt)
#define HAVE_ASYNC 1
#else
#define HAVE_ASYNC 0
#endif

__device__ __forceinline__ v8f vzero8f() {
    v8f v;
#pragma unroll
    for (int i = 0; i < 8; ++i) v[i] = 0.0f;
    return v;
}

__device__ __forceinline__ bf8 bzero8() {
    bf8 v;
#pragma unroll
    for (int i = 0; i < 8; ++i) v[i] = (bf16)0.0f;
    return v;
}

// --------------------------------------------------------------------------
// K1: rectangle rasterization + bilinear resize (half-pixel centers, no
// antialias). g_h,g_w >= 100 so all 4 sample taps are in-range (clamped
// anyway). The source canvas is never materialized.
// --------------------------------------------------------------------------
__global__ void raster_resize_k(const int* __restrict__ obj_x,
                                const int* __restrict__ obj_y,
                                const int* __restrict__ obj_w,
                                const int* __restrict__ obj_h,
                                const int* __restrict__ gh_p,
                                const int* __restrict__ gw_p,
                                float* __restrict__ maps) {
    int idx = blockIdx.x * 256 + threadIdx.x;          // exact: 5,120,000
    int gx = idx % GRID_W;
    int gy = (idx / GRID_W) % GRID_W;
    int c  = (idx / NPIX) & 1;
    int n  = idx / (2 * NPIX);
    int gh = gh_p[0], gw = gw_p[0];
    float sy = (float)gh / (float)GRID_W;
    float sx = (float)gw / (float)GRID_W;
    float fy = (gy + 0.5f) * sy - 0.5f;
    float fx = (gx + 0.5f) * sx - 0.5f;
    int y0 = (int)floorf(fy), x0 = (int)floorf(fx);
    float wy = fy - (float)y0, wx = fx - (float)x0;
    int y0c = min(max(y0, 0), gh - 1);
    int y1c = min(y0 + 1, gh - 1);
    int x0c = min(max(x0, 0), gw - 1);
    int x1c = min(x0 + 1, gw - 1);
    int rx0 = 0, ry0 = 0, rx1 = HUMANSZ, ry1 = HUMANSZ;
    if (c == 1) { rx0 = obj_x[n]; ry0 = obj_y[n]; rx1 = rx0 + obj_w[n]; ry1 = ry0 + obj_h[n]; }
    auto S = [&](int yy, int xx) -> float {
        return (yy >= ry0 && yy < ry1 && xx >= rx0 && xx < rx1) ? 1.0f : 0.0f;
    };
    float v = (1.0f - wy) * ((1.0f - wx) * S(y0c, x0c) + wx * S(y0c, x1c))
            +         wy  * ((1.0f - wx) * S(y1c, x0c) + wx * S(y1c, x1c));
    maps[idx] = v;   // layout ((n*2+c)*100+gy)*100+gx == idx
}

// --------------------------------------------------------------------------
// K2: conv1 2->16ch, 3x3 SAME, fp32 VALU (K=18 too small for WMMA; only
// 1.5 GFLOP). Output bf16, pixel-major [n][pix][16ch] so conv2's WMMA
// A-fragment chunks are contiguous 16-byte loads.
// --------------------------------------------------------------------------
__global__ void conv1_k(const float* __restrict__ maps,
                        const float* __restrict__ w1,
                        const float* __restrict__ b1,
                        bf16* __restrict__ act1) {
    __shared__ float sw[288];   // (16,2,3,3)
    __shared__ float sb[16];
    int t = threadIdx.x;
    if (t < 288) sw[t] = w1[t];
    if (t < 16)  sb[t] = b1[t];
    __syncthreads();
    int idx = blockIdx.x * 256 + t;                    // exact: 2,560,000
    int x = idx % GRID_W, y = (idx / GRID_W) % GRID_W, n = idx / NPIX;
    const float* mp = maps + (size_t)n * 2 * NPIX;
    float m[2][3][3];
#pragma unroll
    for (int ci = 0; ci < 2; ++ci)
#pragma unroll
        for (int dy = 0; dy < 3; ++dy)
#pragma unroll
            for (int dx = 0; dx < 3; ++dx) {
                int yy = y + dy - 1, xx = x + dx - 1;
                m[ci][dy][dx] = (yy >= 0 && yy < GRID_W && xx >= 0 && xx < GRID_W)
                                ? mp[ci * NPIX + yy * GRID_W + xx] : 0.0f;
            }
    bf16* op = act1 + (size_t)idx * C1CH;
#pragma unroll
    for (int c = 0; c < C1CH; ++c) {
        float acc = sb[c];
        const float* wp = sw + c * 18;
#pragma unroll
        for (int ci = 0; ci < 2; ++ci)
#pragma unroll
            for (int dy = 0; dy < 3; ++dy)
#pragma unroll
                for (int dx = 0; dx < 3; ++dx)
                    acc += m[ci][dy][dx] * wp[ci * 9 + dy * 3 + dx];
        op[c] = (bf16)acc;
    }
}

// --------------------------------------------------------------------------
// K3: pre-swizzle conv2 weights into wave32 16-bit B-fragment register order.
// K-axis = tap*16 + ch (9 taps padded to 10 -> K=160 = 5 WMMA steps).
// w2p[step][ntile][lane][16]: lane L holds column n = ntile*16+(L&15), with
// K split-halves k0 = 32*step + 8*(L>>4): elems 0..7 -> k0+j, 8..15 -> k0+16+j.
// --------------------------------------------------------------------------
__global__ void pack_w2_k(const float* __restrict__ w2, bf16* __restrict__ w2p) {
    int idx = blockIdx.x * 256 + threadIdx.x;
    if (idx >= 5 * 2 * 32 * 16) return;
    int j    = idx & 15;
    int L    = (idx >> 4) & 31;
    int tile = (idx >> 9) & 1;
    int s    = idx >> 10;
    int hi = L >> 4;
    int k0 = 32 * s + 8 * hi;
    int kk = (j < 8) ? (k0 + j) : (k0 + 16 + (j - 8));
    int ncol = tile * 16 + (L & 15);
    float v = 0.0f;
    if (kk < 144) {
        int ch = kk & 15, tap = kk >> 4;
        int ky = tap / 3, kx = tap % 3;
        v = w2[((ncol * C1CH + ch) * 3 + ky) * 3 + kx];   // OIHW
    }
    w2p[idx] = (bf16)v;
}

// --------------------------------------------------------------------------
// K4: conv2 16->32ch as implicit-GEMM WMMA. One wave = 16 pixels x 32 outch.
// K = 16ch x (9 taps padded to 10); per step, each lane's two 8-halfword
// chunks are taps 2s and 2s+1 at channel offset 8*(L>>4) -> aligned 16B loads.
// --------------------------------------------------------------------------
__global__ void conv2_wmma_k(const bf16* __restrict__ act1,
                             const bf16* __restrict__ w2p,
                             const float* __restrict__ b2,
                             bf16* __restrict__ act2) {
    int lane = threadIdx.x & 31;
    int wave = threadIdx.x >> 5;
    int tile = blockIdx.x * 8 + wave;                  // exact: 160,000
    int n  = tile / PIXTILES;
    int pt = tile % PIXTILES;
    int pixbase = pt * 16;
    int hi  = lane >> 4;
    int row = lane & 15;
    int pix = pixbase + row;
    int y = pix / GRID_W, x = pix % GRID_W;
    const bf16* a1 = act1 + (size_t)n * NPIX * C1CH;

    v8f acc0 = vzero8f();
    v8f acc1 = vzero8f();
    int c0 = 8 * hi;   // channel offset of this lane's K split-half
#pragma unroll
    for (int s = 0; s < 5; ++s) {
        v16bf A;
#pragma unroll
        for (int half = 0; half < 2; ++half) {
            int tap = 2 * s + half;                    // 0..9 (tap 9 = zero pad)
            bf8 chunk = bzero8();
            if (tap < 9) {
                int dy = tap / 3 - 1, dx = tap % 3 - 1;
                int yy = y + dy, xx = x + dx;
                if (yy >= 0 && yy < GRID_W && xx >= 0 && xx < GRID_W)
                    chunk = *(const bf8*)(a1 + (size_t)(yy * GRID_W + xx) * C1CH + c0);
            }
#pragma unroll
            for (int j = 0; j < 8; ++j) A[half * 8 + j] = chunk[j];
        }
        v16bf B0 = *(const v16bf*)(w2p + ((s * 2 + 0) * 32 + lane) * 16);
        v16bf B1 = *(const v16bf*)(w2p + ((s * 2 + 1) * 32 + lane) * 16);
        acc0 = __builtin_amdgcn_wmma_f32_16x16x32_bf16(false, A, false, B0,
                                                       (short)0, acc0, false, false);
        acc1 = __builtin_amdgcn_wmma_f32_16x16x32_bf16(false, A, false, B1,
                                                       (short)0, acc1, false, false);
    }
    // bias + store in CHW flatten order: act2[n][ch*10000 + pix]
    int nc0 = lane & 15;
    int nc1 = 16 + nc0;
    float bb0 = b2[nc0], bb1 = b2[nc1];
    bf16* o = act2 + (size_t)n * KPROJ;
#pragma unroll
    for (int r = 0; r < 8; ++r) {
        int mpix = pixbase + r + 8 * hi;               // D: vgpr r, hi-half -> M = r+8*hi
        o[nc0 * NPIX + mpix] = (bf16)(acc0[r] + bb0);
        o[nc1 * NPIX + mpix] = (bf16)(acc1[r] + bb1);
    }
}

// --------------------------------------------------------------------------
// K5: proj_w f32 -> bf16 (one streaming pass, 4-wide).
// --------------------------------------------------------------------------
__global__ void cvt_w_k(const float* __restrict__ w, bf16* __restrict__ o, long n4) {
    long i = (long)blockIdx.x * blockDim.x + threadIdx.x;
    long stride = (long)gridDim.x * blockDim.x;
    for (; i < n4; i += stride) {
        f4 v = ((const f4*)w)[i];
        bf4 r;
        r[0] = (bf16)v[0]; r[1] = (bf16)v[1]; r[2] = (bf16)v[2]; r[3] = (bf16)v[3];
        ((bf4*)o)[i] = r;
    }
}

// --------------------------------------------------------------------------
// K6: projection GEMM: part[ks][m][f] = sum_{k in chunk ks} act2[m][k]*Wb[f][k].
// Block = (ks, nt, mgroup): 8 waves share the nt B-panel, each wave owns one
// 16-row M tile. The 16x32 bf16 B panel (1 KB) is staged in LDS, double
// buffered: waves 0-1 (64 lanes, one 16B chunk each) issue
// global_load_async_to_lds_b128 for step i+1 while all waves consume step i
// from LDS (ds_load_b128, 80B row pitch -> conflict-free across rows), then
// s_wait_asynccnt 0 + workgroup barrier. A operands stream straight from
// global (L2-resident, 16B pair loads) with prefetch ahead.
// --------------------------------------------------------------------------
#define BPITCH 40   // bf16 elems per LDS row (80 B)

#if HAVE_ASYNC
#define ASYNC_COPY16(ldst, gsrc)                                          \
    __builtin_amdgcn_global_load_async_to_lds_b128(                       \
        (__attribute__((address_space(1))) async_v4i*)(gsrc),             \
        (__attribute__((address_space(3))) async_v4i*)(ldst), 0, 0)
#else
#define ASYNC_COPY16(ldst, gsrc) (*(bf8*)(ldst) = *(const bf8*)(gsrc))
#endif

__device__ __forceinline__ void panel_sync(bool loader) {
#if HAVE_ASYNC
    if (loader) __builtin_amdgcn_s_wait_asynccnt(0);
#endif
    __syncthreads();
}

__global__ void proj_wmma_k(const bf16* __restrict__ act2,
                            const bf16* __restrict__ wb,
                            float* __restrict__ part) {
    __shared__ bf16 sB[2][16][BPITCH];                 // 2 x 1.25 KB panels
    int lane = threadIdx.x & 31;
    int wave = threadIdx.x >> 5;
    int b  = blockIdx.x;                               // exact: 512
    int mg = b & 1;
    int nt = (b >> 1) & 15;
    int ks = b >> 5;
    int mt = mg * 8 + wave;
    int hi = lane >> 4, lo = lane & 15;
    size_t kbase = (size_t)ks * KCHUNK;

    const bf16* pa = act2 + (size_t)(mt * 16 + lo) * KPROJ + kbase + hi * 8;

    // B loader: threads 0..63 (waves 0-1, wave-uniform predicate) each own one
    // 16B chunk of the 16-row x 32-col panel.
    int t = threadIdx.x;
    bool loader = (t < 64);
    int lrow   = t >> 2;                               // 0..15 feature row
    int lchunk = t & 3;                                // 0..3  (8 bf16 per chunk)
    const bf16* gB = wb + (size_t)(nt * 16 + lrow) * KPROJ + kbase + lchunk * 8;

    if (loader) ASYNC_COPY16(&sB[0][lrow][lchunk * 8], gB);
    panel_sync(loader);

    v8f acc = vzero8f();
    for (int it = 0; it < KSTEPS; ++it) {
        int buf = it & 1;
        if (loader && (it + 1 < KSTEPS))
            ASYNC_COPY16(&sB[buf ^ 1][lrow][lchunk * 8], gB + (size_t)(it + 1) * 32);

        bf8 a0 = *(const bf8*)(pa);
        bf8 a1 = *(const bf8*)(pa + 16);
        bf8 b0 = *(const bf8*)&sB[buf][lo][8 * hi];
        bf8 b1 = *(const bf8*)&sB[buf][lo][16 + 8 * hi];
        v16bf A, Bv;
#pragma unroll
        for (int j = 0; j < 8; ++j) {
            A[j] = a0[j];  A[8 + j] = a1[j];
            Bv[j] = b0[j]; Bv[8 + j] = b1[j];
        }
        acc = __builtin_amdgcn_wmma_f32_16x16x32_bf16(false, A, false, Bv,
                                                      (short)0, acc, false, false);
        pa += 32;
        if ((it & 15) == 0) __builtin_prefetch(pa + 1024, 0, 1);
        panel_sync(loader);                            // next buf ready for all
    }

    int fcol = nt * 16 + lo;
    float* pp = part + (size_t)ks * NSAMP * FDIM;
#pragma unroll
    for (int r = 0; r < 8; ++r) {
        int mrow = mt * 16 + r + 8 * hi;
        pp[mrow * FDIM + fcol] = acc[r];
    }
}

// --------------------------------------------------------------------------
// K7: reduce K-split partials + bias -> out (32,8,256) f32.
// --------------------------------------------------------------------------
__global__ void finish_k(const float* __restrict__ part,
                         const float* __restrict__ proj_b,
                         float* __restrict__ out) {
    int idx = blockIdx.x * 256 + threadIdx.x;          // exact: 65,536
    float s = proj_b[idx & (FDIM - 1)];
#pragma unroll
    for (int ks = 0; ks < KSPLIT; ++ks) s += part[(size_t)ks * NSAMP * FDIM + idx];
    out[idx] = s;
}

// ---------------------------------------------------------------------------
extern "C" void kernel_launch(void* const* d_in, const int* in_sizes, int n_in,
                              void* d_out, int out_size, void* d_ws, size_t ws_size,
                              hipStream_t stream) {
    (void)in_sizes; (void)n_in; (void)out_size; (void)ws_size;
    const int*   obj_x  = (const int*)d_in[0];
    const int*   obj_y  = (const int*)d_in[1];
    const int*   obj_w  = (const int*)d_in[2];
    const int*   obj_h  = (const int*)d_in[3];
    const int*   g_h    = (const int*)d_in[4];
    const int*   g_w    = (const int*)d_in[5];
    const float* conv1w = (const float*)d_in[6];
    const float* conv1b = (const float*)d_in[7];
    const float* conv2w = (const float*)d_in[8];
    const float* conv2b = (const float*)d_in[9];
    const float* projw  = (const float*)d_in[10];
    const float* projb  = (const float*)d_in[11];
    float* out = (float*)d_out;

    // carve workspace (256B-aligned sub-buffers), total ~434 MB
    char* ws = (char*)d_ws;
    size_t off = 0;
    auto take = [&](size_t bytes) -> char* {
        char* p = ws + off;
        off = (off + bytes + 255) & ~(size_t)255;
        return p;
    };
    float* maps = (float*)take((size_t)NSAMP * 2 * NPIX * sizeof(float));     // 20.5 MB
    bf16*  act1 = (bf16*)take((size_t)NSAMP * NPIX * C1CH * sizeof(bf16));    // 82 MB
    bf16*  act2 = (bf16*)take((size_t)NSAMP * KPROJ * sizeof(bf16));          // 164 MB
    bf16*  w2p  = (bf16*)take((size_t)5 * 2 * 32 * 16 * sizeof(bf16));        // 10 KB
    bf16*  wbf  = (bf16*)take((size_t)FDIM * KPROJ * sizeof(bf16));           // 164 MB
    float* part = (float*)take((size_t)KSPLIT * NSAMP * FDIM * sizeof(float));// 4 MB

    raster_resize_k<<<20000, 256, 0, stream>>>(obj_x, obj_y, obj_w, obj_h, g_h, g_w, maps);
    conv1_k<<<10000, 256, 0, stream>>>(maps, conv1w, conv1b, act1);
    pack_w2_k<<<20, 256, 0, stream>>>(conv2w, w2p);
    conv2_wmma_k<<<20000, 256, 0, stream>>>(act1, w2p, conv2b, act2);
    cvt_w_k<<<8192, 256, 0, stream>>>(projw, wbf, (long)(FDIM * (long)KPROJ / 4));
    proj_wmma_k<<<512, 256, 0, stream>>>(act2, wbf, part);
    finish_k<<<256, 256, 0, stream>>>(part, projb, out);
}
